// CoarseMatching_9405978378740
// MI455X (gfx1250) — matrix operand order repaired and checked by reading, not compiled, not run on hardware
//
#include <hip/hip_runtime.h>
#include <hip/hip_bf16.h>
#include <cstdint>

#define NB    2
#define CC    256
#define GHW   40
#define G16   1600
#define LFINE 6400
#define KK4   32
#define TOPK  8

typedef __attribute__((ext_vector_type(16))) __bf16 v16bf;
typedef __attribute__((ext_vector_type(8)))  __bf16 v8bf;
typedef __attribute__((ext_vector_type(8)))  float  v8f;

// ---------------------------------------------------------------- priors -> bf16 (N, G, C) row-major
__global__ void k_tp(const float* __restrict__ prior, __bf16* __restrict__ pt) {
  int t = blockIdx.x * blockDim.x + threadIdx.x;
  if (t >= NB * G16 * CC) return;
  int n = t / (G16 * CC);
  int rem = t % (G16 * CC);
  int i = rem / CC, c = rem % CC;
  pt[t] = (__bf16)prior[(size_t)n * CC * G16 + (size_t)c * G16 + i];
}

// ---------------------------------------------------------------- sim = p0^T p1 / C  via bf16 WMMA
__global__ __launch_bounds__(256) void k_sim(const __bf16* __restrict__ p0t,
                                             const __bf16* __restrict__ p1t,
                                             float* __restrict__ sim) {
  int wave = blockIdx.x * (blockDim.x >> 5) + (threadIdx.x >> 5);
  if (wave >= NB * 100 * 100) return;
  int n  = wave / (100 * 100);
  int t  = wave % (100 * 100);
  int ti = t / 100, tj = t % 100;
  int lane = threadIdx.x & 31;
  int half = lane >> 4, idx = lane & 15;
  const __bf16* A = p0t + ((size_t)n * G16 + ti * 16 + idx) * CC;
  const __bf16* B = p1t + ((size_t)n * G16 + tj * 16 + idx) * CC;
  v8f acc = {0.f, 0.f, 0.f, 0.f, 0.f, 0.f, 0.f, 0.f};
  for (int k0 = 0; k0 < CC; k0 += 32) {
    __builtin_prefetch(A + k0 + 64, 0, 0);
    __builtin_prefetch(B + k0 + 64, 0, 0);
    // A fragment: lane<16 -> K {0..7,16..23}, lane>=16 -> K {8..15,24..31}
    v8bf alo = *(const v8bf*)(A + k0 + 8 * half);
    v8bf ahi = *(const v8bf*)(A + k0 + 16 + 8 * half);
    // B fragment: lane holds column (idx), lane<16 -> K {0..7,16..23}, lane>=16 -> K {8..15,24..31}
    v8bf blo = *(const v8bf*)(B + k0 + 8 * half);
    v8bf bhi = *(const v8bf*)(B + k0 + 16 + 8 * half);
    v16bf af  = __builtin_shufflevector(alo, ahi, 0,1,2,3,4,5,6,7,8,9,10,11,12,13,14,15);
    v16bf bfr = __builtin_shufflevector(blo, bhi, 0,1,2,3,4,5,6,7,8,9,10,11,12,13,14,15);
    acc = __builtin_amdgcn_wmma_f32_16x16x32_bf16(false, af, false, bfr, (short)0, acc, false, false);
  }
  const float sc = 1.0f / 256.0f;
  size_t base = (size_t)n * G16 * G16;
  for (int r = 0; r < 8; r++) {
    int m   = ti * 16 + r + 8 * half;   // D layout: M = r + 8*half
    int col = tj * 16 + idx;            // N = lane&15
    sim[base + (size_t)m * G16 + col] = acc[r] * sc;
  }
}

// ---------------------------------------------------------------- top-8 per row (or column) of sim
__global__ void k_topk(const float* __restrict__ sim, int* __restrict__ iout, int transposed) {
  int wave = blockIdx.x * (blockDim.x >> 5) + (threadIdx.x >> 5);
  int lane = threadIdx.x & 31;
  if (wave >= NB * G16) return;
  int n = wave / G16, i = wave % G16;
  const float* row = sim + (size_t)n * G16 * G16 + (transposed ? (size_t)i : (size_t)i * G16);
  size_t stride = transposed ? (size_t)G16 : 1;
  int sel[TOPK];
  for (int p = 0; p < TOPK; p++) {
    float best = -3.4e38f;
    int bidx = 0x7fffffff;
    for (int j = lane; j < G16; j += 32) {
      bool skip = false;
      for (int q = 0; q < p; q++) skip |= (sel[q] == j);
      float v = row[(size_t)j * stride];
      if (!skip && (v > best || (v == best && j < bidx))) { best = v; bidx = j; }
    }
    for (int off = 16; off; off >>= 1) {
      float ov = __shfl_xor(best, off, 32);
      int   oi = __shfl_xor(bidx, off, 32);
      if (ov > best || (ov == best && oi < bidx)) { best = ov; bidx = oi; }
    }
    sel[p] = bidx;
    if (lane == 0) iout[(size_t)wave * TOPK + p] = bidx;
  }
}

// ---------------------------------------------------------------- fine index maps
__global__ void k_map(const int* __restrict__ i01, const int* __restrict__ i10,
                      int* __restrict__ ind0, int* __restrict__ ind1) {
  int t = blockIdx.x * blockDim.x + threadIdx.x;
  if (t >= NB * LFINE * KK4) return;
  int n = t / (LFINE * KK4);
  int rem = t % (LFINE * KK4);
  int l = rem / KK4, k = rem % KK4;
  int r = l / 80, c = l % 80;
  int g  = (r >> 1) * GHW + (c >> 1);
  int e  = k & 3, tk = k >> 2;
  {
    int idx = i01[((size_t)n * G16 + g) * TOPK + tk];
    int rr = (idx / GHW) * 2 + (e >> 1);
    int cc = (idx % GHW) * 2 + (e & 1);
    ind0[t] = rr * 80 + cc;
  }
  {
    int idx = i10[((size_t)n * G16 + g) * TOPK + tk];
    int rr = (idx / GHW) * 2 + (e >> 1);
    int cc = (idx % GHW) * 2 + (e & 1);
    ind1[(size_t)n * KK4 * LFINE + (size_t)k * LFINE + l] = rr * 80 + cc;
  }
}

// ---------------------------------------------------------------- heatmap einsum + softmax (WMMA)
// dir 0: s01 = x0 . a01^T ;  dir 1: s10 = x1 . a10^T  (A = attended 32xC, B = x 4xC)
__global__ __launch_bounds__(32) void k_heat(const float* __restrict__ x0, const float* __restrict__ x1,
                                             const float* __restrict__ a01, const float* __restrict__ a10,
                                             float* __restrict__ h01p, float* __restrict__ h10p) {
  int bid = blockIdx.x;
  int dir = bid >= NB * G16;
  int gl  = dir ? bid - NB * G16 : bid;
  int n = gl / G16, g = gl % G16;
  const float* X = (dir ? x1 : x0) + ((size_t)n * G16 + g) * 4 * CC;
  const float* A = (dir ? a10 : a01) + ((size_t)n * G16 + g) * KK4 * CC;
  int lane = threadIdx.x;
  int half = lane >> 4, idx = lane & 15;
  int qc = idx < 4 ? idx : 0;  // clamp B column loads (only q=0..3 valid)
  v8f acc0 = {0.f,0.f,0.f,0.f,0.f,0.f,0.f,0.f};
  v8f acc1 = acc0;
  for (int c0 = 0; c0 < CC; c0 += 32) {
    v16bf af0, af1, bfr;
    const float* a0p = A + (size_t)idx * CC + c0;
    const float* a1p = A + (size_t)(16 + idx) * CC + c0;
    for (int e = 0; e < 8; e++) {
      af0[e]     = (__bf16)a0p[8 * half + e];
      af0[8 + e] = (__bf16)a0p[16 + 8 * half + e];
      af1[e]     = (__bf16)a1p[8 * half + e];
      af1[8 + e] = (__bf16)a1p[16 + 8 * half + e];
    }
    const float* bp = X + (size_t)qc * CC + c0 + 8 * half;
    for (int e = 0; e < 8; e++) { bfr[e] = (__bf16)bp[e]; bfr[8 + e] = (__bf16)bp[16 + e]; }
    acc0 = __builtin_amdgcn_wmma_f32_16x16x32_bf16(false, af0, false, bfr, (short)0, acc0, false, false);
    acc1 = __builtin_amdgcn_wmma_f32_16x16x32_bf16(false, af1, false, bfr, (short)0, acc1, false, false);
  }
  __shared__ float s[KK4][4];
  const float sc = 1.0f / (256.0f * 0.1f);  // 1/C then 1/TEMP
  for (int r = 0; r < 8; r++) {
    if (idx < 4) {
      s[r + 8 * half][idx]      = acc0[r] * sc;
      s[16 + r + 8 * half][idx] = acc1[r] * sc;
    }
  }
  __syncthreads();
  if (lane < 4) {
    int q = lane;
    float m = -3.4e38f;
    for (int k = 0; k < KK4; k++) m = fmaxf(m, s[k][q]);
    float ev[KK4], sum = 0.f;
    for (int k = 0; k < KK4; k++) { ev[k] = __expf(s[k][q] - m); sum += ev[k]; }
    float inv = 1.0f / sum;
    int gr = g / GHW, gc = g % GHW;
    int sr = q >> 1, scq = q & 1;
    int lpix = (gr * 2 + sr) * 80 + gc * 2 + scq;
    if (!dir) {
      float* o = h01p + ((size_t)n * LFINE + lpix) * KK4;
      for (int k = 0; k < KK4; k++) o[k] = ev[k] * inv;
    } else {
      float* o = h10p + (size_t)n * KK4 * LFINE + lpix;
      for (int k = 0; k < KK4; k++) o[(size_t)k * LFINE] = ev[k] * inv;
    }
  }
}

// ---------------------------------------------------------------- mutual fusion (sparse form of scatter/gather)
__global__ void k_fuse01(const float* __restrict__ h01p, const float* __restrict__ h10p,
                         const int* __restrict__ ind0, const int* __restrict__ ind1,
                         float* __restrict__ o_h01) {
  int t = blockIdx.x * blockDim.x + threadIdx.x;
  if (t >= NB * LFINE * KK4) return;
  int n = t / (LFINE * KK4);
  int rem = t % (LFINE * KK4);
  int l0 = rem / KK4;
  int j = ind0[t];
  const int*   col = ind1 + (size_t)n * KK4 * LFINE;
  const float* h10 = h10p + (size_t)n * KK4 * LFINE;
  float z = 0.f;
  for (int kk = 0; kk < KK4; kk++) {
    if (col[(size_t)kk * LFINE + j] == l0) { z = h10[(size_t)kk * LFINE + j]; break; }
  }
  o_h01[t] = h01p[t] * z;
}

__global__ void k_fuse10(const float* __restrict__ h01p, const float* __restrict__ h10p,
                         const int* __restrict__ ind0, const int* __restrict__ ind1,
                         float* __restrict__ o_h10) {
  int t = blockIdx.x * blockDim.x + threadIdx.x;
  if (t >= NB * KK4 * LFINE) return;
  int n = t / (KK4 * LFINE);
  int rem = t % (KK4 * LFINE);
  int l1 = rem % LFINE;
  int i = ind1[t];
  const int*   rowi = ind0 + ((size_t)n * LFINE + i) * KK4;
  const float* h01  = h01p + ((size_t)n * LFINE + i) * KK4;
  float z = 0.f;
  for (int k2 = 0; k2 < KK4; k2++) {
    if (rowi[k2] == l1) { z = h01[k2]; break; }
  }
  o_h10[t] = h10p[t] * z;
}

// ---------------------------------------------------------------- coarse matching
__global__ void k_match0(const float* __restrict__ h01, const int* __restrict__ ind0,
                         float* __restrict__ score_ws, int* __restrict__ jb_ws,
                         float* __restrict__ o_jb) {
  int t = blockIdx.x * blockDim.x + threadIdx.x;
  if (t >= NB * LFINE) return;
  int l0 = t % LFINE;
  const float* row = h01 + (size_t)t * KK4;
  float best = row[0]; int bk = 0;
  for (int k = 1; k < KK4; k++) if (row[k] > best) { best = row[k]; bk = k; }
  int j = ind0[(size_t)t * KK4 + bk];
  int r = l0 / 80, c = l0 % 80;
  if (r < 2 || r >= 78 || c < 2 || c >= 78) j = 0;
  score_ws[t] = best;
  jb_ws[t] = j;
  o_jb[t] = (float)j;
}

__global__ void k_match1(const float* __restrict__ h10, const int* __restrict__ ind1,
                         int* __restrict__ ib_ws) {
  int t = blockIdx.x * blockDim.x + threadIdx.x;
  if (t >= NB * LFINE) return;
  int n = t / LFINE, l1 = t % LFINE;
  const float* col = h10 + (size_t)n * KK4 * LFINE + l1;
  float best = col[0]; int bk = 0;
  for (int k = 1; k < KK4; k++) {
    float v = col[(size_t)k * LFINE];
    if (v > best) { best = v; bk = k; }
  }
  int i = ind1[(size_t)n * KK4 * LFINE + (size_t)bk * LFINE + l1];
  int r = l1 / 80, c = l1 % 80;
  if (r < 2 || r >= 78 || c < 2 || c >= 78) i = 0;
  ib_ws[t] = i;
}

__global__ void k_final(const float* __restrict__ score_ws, const int* __restrict__ jb_ws,
                        const int* __restrict__ ib_ws, float* __restrict__ o_sc,
                        float* __restrict__ o_mk) {
  int t = blockIdx.x * blockDim.x + threadIdx.x;
  if (t >= NB * LFINE) return;
  int n = t / LFINE, l0 = t % LFINE;
  int j = jb_ws[t];
  int bip = ib_ws[(size_t)n * LFINE + j];
  float sc = score_ws[t];
  bool m = (bip == l0) && (l0 != 0) && (sc > 0.2f);
  o_mk[t] = m ? 1.0f : 0.0f;
  o_sc[t] = m ? sc : 0.0f;
}

// ---------------------------------------------------------------- unpatchify copies
__global__ void k_unpatch(const float* __restrict__ x, float* __restrict__ out) {
  int t = blockIdx.x * blockDim.x + threadIdx.x;
  if (t >= NB * CC * LFINE) return;
  int n = t / (CC * LFINE);
  int rem = t % (CC * LFINE);
  int c = rem / LFINE, pix = rem % LFINE;
  int r = pix / 80, w = pix % 80;
  int gr = r >> 1, sr = r & 1, gc = w >> 1, sc = w & 1;
  out[t] = x[(((size_t)n * G16 + gr * GHW + gc) * 4 + (sr * 2 + sc)) * CC + c];
}

// ================================================================ launch
extern "C" void kernel_launch(void* const* d_in, const int* in_sizes, int n_in,
                              void* d_out, int out_size, void* d_ws, size_t ws_size,
                              hipStream_t stream) {
  const float* x0  = (const float*)d_in[0];
  const float* x1  = (const float*)d_in[1];
  const float* a01 = (const float*)d_in[2];
  const float* a10 = (const float*)d_in[3];
  const float* p0  = (const float*)d_in[4];
  const float* p1  = (const float*)d_in[5];

  char* w = (char*)d_ws;
  __bf16* p0t = (__bf16*)w; w += (size_t)NB * G16 * CC * sizeof(__bf16);
  __bf16* p1t = (__bf16*)w; w += (size_t)NB * G16 * CC * sizeof(__bf16);
  float*  sim = (float*)w;  w += (size_t)NB * G16 * G16 * sizeof(float);
  int*    i01 = (int*)w;    w += (size_t)NB * G16 * TOPK * sizeof(int);
  int*    i10 = (int*)w;    w += (size_t)NB * G16 * TOPK * sizeof(int);
  int*   ind0 = (int*)w;    w += (size_t)NB * LFINE * KK4 * sizeof(int);
  int*   ind1 = (int*)w;    w += (size_t)NB * KK4 * LFINE * sizeof(int);
  float* h01p = (float*)w;  w += (size_t)NB * LFINE * KK4 * sizeof(float);
  float* h10p = (float*)w;  w += (size_t)NB * KK4 * LFINE * sizeof(float);
  float* score_ws = (float*)w; w += (size_t)NB * LFINE * sizeof(float);
  int*   jb_ws = (int*)w;   w += (size_t)NB * LFINE * sizeof(int);
  int*   ib_ws = (int*)w;   w += (size_t)NB * LFINE * sizeof(int);

  float* out   = (float*)d_out;
  float* o_h01 = out;
  float* o_h10 = o_h01 + (size_t)NB * LFINE * KK4;
  float* o_sc  = o_h10 + (size_t)NB * KK4 * LFINE;
  float* o_jb  = o_sc  + (size_t)NB * LFINE;
  float* o_mk  = o_jb  + (size_t)NB * LFINE;
  float* o_x08 = o_mk  + (size_t)NB * LFINE;
  float* o_x18 = o_x08 + (size_t)NB * CC * LFINE;

  {
    int total = NB * G16 * CC;
    k_tp<<<(total + 255) / 256, 256, 0, stream>>>(p0, p0t);
    k_tp<<<(total + 255) / 256, 256, 0, stream>>>(p1, p1t);
  }
  k_sim<<<(NB * 100 * 100 + 7) / 8, 256, 0, stream>>>(p0t, p1t, sim);
  k_topk<<<(NB * G16 + 3) / 4, 128, 0, stream>>>(sim, i01, 0);
  k_topk<<<(NB * G16 + 3) / 4, 128, 0, stream>>>(sim, i10, 1);
  {
    int total = NB * LFINE * KK4;
    k_map<<<(total + 255) / 256, 256, 0, stream>>>(i01, i10, ind0, ind1);
  }
  k_heat<<<2 * NB * G16, 32, 0, stream>>>(x0, x1, a01, a10, h01p, h10p);
  {
    int total = NB * LFINE * KK4;
    k_fuse01<<<(total + 255) / 256, 256, 0, stream>>>(h01p, h10p, ind0, ind1, o_h01);
    k_fuse10<<<(total + 255) / 256, 256, 0, stream>>>(h01p, h10p, ind0, ind1, o_h10);
  }
  {
    int total = NB * LFINE;
    k_match0<<<(total + 255) / 256, 256, 0, stream>>>(o_h01, ind0, score_ws, jb_ws, o_jb);
    k_match1<<<(total + 255) / 256, 256, 0, stream>>>(o_h10, ind1, ib_ws);
    k_final<<<(total + 255) / 256, 256, 0, stream>>>(score_ws, jb_ws, ib_ws, o_sc, o_mk);
  }
  {
    int total = NB * CC * LFINE;
    k_unpatch<<<(total + 255) / 256, 256, 0, stream>>>(x0, o_x08);
    k_unpatch<<<(total + 255) / 256, 256, 0, stream>>>(x1, o_x18);
  }
}
